// SNN_25323127177359
// MI455X (gfx1250) — matrix operand order repaired and checked by reading, not compiled, not run on hardware
//
#include <hip/hip_runtime.h>

typedef __attribute__((ext_vector_type(16))) __bf16 v16bf;
typedef __attribute__((ext_vector_type(8)))  float  v8f;

#define N_DIM 2048
#define B_DIM 512
#define BM 64
#define BN 128
#define BK 32
#define LDSB 40   // padded LDS row stride in bf16 elements (80B rows, 16B aligned sublocks)

struct LayerArgs {
  const void*  A[5];
  const float* W[5];
  const float* M[5];
  const float* bias;
  float        biasScale;
  __bf16*      out;
  int          numPreds;
  int          K;
};

union Frag  { uint4 u[2]; v16bf v; };
union Pack4 { __bf16 h[4]; uint2 u; };
union Pack8 { __bf16 h[8]; uint4 u; };

template<bool A_BF16, bool HAS_MASK, bool K_ALIGNED>
__global__ __launch_bounds__(256)
void dag_layer_kernel(LayerArgs args) {
  __shared__ __align__(16) __bf16 smA[2][BM * LDSB];
  __shared__ __align__(16) __bf16 smB[2][BN * LDSB];

  const int tid       = threadIdx.x;
  const int nBase     = blockIdx.x * BN;     // output-feature tile
  const int batchBase = blockIdx.y * BM;     // batch tile
  const int K         = args.K;
  const int nChunks   = (K + BK - 1) / BK;

  const int lane = tid & 31;
  const int wave = tid >> 5;
  const int wm   = wave >> 2;    // 0..1 : 32-row slab within block tile
  const int wn   = wave & 3;     // 0..3 : 32-col slab
  const int lr   = lane & 15;
  const int lh   = lane >> 4;

  v8f acc[2][2] = {};

  for (int j = 0; j < args.numPreds; ++j) {
    const float* Wp = args.W[j];
    const float* Mp = args.M[j];
    const void*  Ap = args.A[j];

    auto load_tiles = [&](int buf, int kBase) {
      // ---- B tile: BN rows of masked weight (row-major => B-matrix columns) ----
      const int rB = tid >> 3;           // 0..31
      const int k4 = (tid & 7) * 4;
      #pragma unroll
      for (int p = 0; p < 4; ++p) {
        const int    r   = rB + p * 32;  // 0..127
        const size_t off = (size_t)(nBase + r) * K + kBase + k4;
        float4 w;
        if constexpr (K_ALIGNED) {
          w = *(const float4*)(Wp + off);
          if constexpr (HAS_MASK) {
            float4 m = *(const float4*)(Mp + off);
            w.x *= m.x; w.y *= m.y; w.z *= m.z; w.w *= m.w;
          }
          __builtin_prefetch(Wp + off + BK, 0, 3);           // next k-chunk -> near cache
          if constexpr (HAS_MASK) __builtin_prefetch(Mp + off + BK, 0, 3);
        } else {
          w = make_float4(0.f, 0.f, 0.f, 0.f);
          if (kBase + k4 < K) {
            w = *(const float4*)(Wp + off);
            if constexpr (HAS_MASK) {
              float4 m = *(const float4*)(Mp + off);
              w.x *= m.x; w.y *= m.y; w.z *= m.z; w.w *= m.w;
            }
            if (kBase + BK < K) __builtin_prefetch(Wp + off + BK, 0, 3);
          }
        }
        Pack4 pk;
        pk.h[0] = (__bf16)w.x; pk.h[1] = (__bf16)w.y;
        pk.h[2] = (__bf16)w.z; pk.h[3] = (__bf16)w.w;
        *(uint2*)&smB[buf][r * LDSB + k4] = pk.u;
      }
      // ---- A tile: BM rows of activations ----
      const int rA = tid >> 2;           // 0..63
      const int c8 = (tid & 3) * 8;
      if constexpr (A_BF16) {
        const __bf16* a = (const __bf16*)Ap + (size_t)(batchBase + rA) * K + kBase + c8;
        *(uint4*)&smA[buf][rA * LDSB + c8] = *(const uint4*)a;   // K % 32 == 0 here
      } else {
        const float* a = (const float*)Ap + (size_t)(batchBase + rA) * K + kBase + c8;
        float4 lo = make_float4(0.f, 0.f, 0.f, 0.f), hi = lo;
        if (K_ALIGNED || kBase + c8     < K) lo = *(const float4*)a;
        if (K_ALIGNED || kBase + c8 + 4 < K) hi = *(const float4*)(a + 4);
        Pack8 pk;
        pk.h[0]=(__bf16)lo.x; pk.h[1]=(__bf16)lo.y; pk.h[2]=(__bf16)lo.z; pk.h[3]=(__bf16)lo.w;
        pk.h[4]=(__bf16)hi.x; pk.h[5]=(__bf16)hi.y; pk.h[6]=(__bf16)hi.z; pk.h[7]=(__bf16)hi.w;
        *(uint4*)&smA[buf][rA * LDSB + c8] = pk.u;
      }
    };

    __syncthreads();            // protect buffer 0 from previous predecessor's last compute
    load_tiles(0, 0);
    int buf = 0;
    for (int kc = 0; kc < nChunks; ++kc) {
      __syncthreads();
      if (kc + 1 < nChunks) load_tiles(buf ^ 1, (kc + 1) * BK);

      // A fragment: 16x32 bf16. lane<16: K0-7 then K16-23; lane>=16: K8-15 then K24-31.
      Frag a[2], b[2];
      #pragma unroll
      for (int fm = 0; fm < 2; ++fm) {
        const __bf16* p = &smA[buf][(wm * 32 + fm * 16 + lr) * LDSB + lh * 8];
        a[fm].u[0] = *(const uint4*)p;          // K = lh*8 .. +7
        a[fm].u[1] = *(const uint4*)(p + 16);   // K = 16 + lh*8 .. +7
      }
      // B fragment: 32x16 bf16; column = Wm row; lane<16: K0-7 then K16-23; lane>=16: K8-15 then K24-31.
      #pragma unroll
      for (int fn = 0; fn < 2; ++fn) {
        const __bf16* p = &smB[buf][(wn * 32 + fn * 16 + lr) * LDSB + lh * 8];
        b[fn].u[0] = *(const uint4*)p;
        b[fn].u[1] = *(const uint4*)(p + 16);
      }
      #pragma unroll
      for (int fm = 0; fm < 2; ++fm)
        #pragma unroll
        for (int fn = 0; fn < 2; ++fn)
          acc[fm][fn] = __builtin_amdgcn_wmma_f32_16x16x32_bf16(
              false, a[fm].v, false, b[fn].v, (short)0, acc[fm][fn], false, false);
      buf ^= 1;
    }
  }

  // Epilogue: bias (scaled) + ReLU, store bf16 activation.
  // C layout: VGPR r -> M = r + 8*(lane>=16); N = lane & 15.
  #pragma unroll
  for (int fm = 0; fm < 2; ++fm) {
    #pragma unroll
    for (int fn = 0; fn < 2; ++fn) {
      const int   colG = nBase + wn * 32 + fn * 16 + lr;
      const float bv   = args.biasScale * args.bias[colG];
      #pragma unroll
      for (int r = 0; r < 8; ++r) {
        const int rowG = batchBase + wm * 32 + fm * 16 + r + lh * 8;
        float v = acc[fm][fn][r] + bv;
        v = v > 0.f ? v : 0.f;
        args.out[(size_t)rowG * N_DIM + colG] = (__bf16)v;
      }
    }
  }
}

// out[b,c] = act5[b]·w1[c] + b1[c] + act4[b]·w2[c] + b2[c]   (512x10, trivial)
__global__ __launch_bounds__(256)
void classifier_kernel(const __bf16* __restrict__ a5, const __bf16* __restrict__ a4,
                       const float* __restrict__ w1, const float* __restrict__ b1,
                       const float* __restrict__ w2, const float* __restrict__ b2,
                       float* __restrict__ out) {
  const int bidx = blockIdx.x;
  const int t    = threadIdx.x;
  const __bf16* r5 = a5 + (size_t)bidx * N_DIM;
  const __bf16* r4 = a4 + (size_t)bidx * N_DIM;
  float s[10];
  #pragma unroll
  for (int c = 0; c < 10; ++c) s[c] = 0.f;
  for (int n = t; n < N_DIM; n += 256) {
    const float x5 = (float)r5[n];
    const float x4 = (float)r4[n];
    #pragma unroll
    for (int c = 0; c < 10; ++c)
      s[c] += x5 * w1[c * N_DIM + n] + x4 * w2[c * N_DIM + n];
  }
  __shared__ float red[256];
  for (int c = 0; c < 10; ++c) {
    red[t] = s[c];
    __syncthreads();
    for (int off = 128; off > 0; off >>= 1) {
      if (t < off) red[t] += red[t + off];
      __syncthreads();
    }
    if (t == 0) out[bidx * 10 + c] = red[0] + b1[c] + b2[c];
    __syncthreads();
  }
}

extern "C" void kernel_launch(void* const* d_in, const int* in_sizes, int n_in,
                              void* d_out, int out_size, void* d_ws, size_t ws_size,
                              hipStream_t stream) {
  const float* x     = (const float*)d_in[0];
  const float* w_in  = (const float*)d_in[1];
  const float* b_in  = (const float*)d_in[2];
  const float* W     = (const float*)d_in[3];
  const float* M     = (const float*)d_in[4];
  const float* bvec  = (const float*)d_in[5];
  const float* w1    = (const float*)d_in[6];
  const float* b1    = (const float*)d_in[7];
  const float* w2    = (const float*)d_in[8];
  const float* b2    = (const float*)d_in[9];
  float* out = (float*)d_out;

  __bf16* acts = (__bf16*)d_ws;  // 6 * 512 * 2048 bf16 = 12 MB
  auto act = [&](int i) { return acts + (size_t)i * B_DIM * N_DIM; };

  const dim3 grid(N_DIM / BN, B_DIM / BM);  // 16 x 8
  const dim3 blk(256);

  // Input layer: relu(x @ w_in.T + b_in), K = 784 (not BK-aligned), fp32 A, no mask.
  {
    LayerArgs la{};
    la.A[0] = x; la.W[0] = w_in; la.M[0] = nullptr;
    la.bias = b_in; la.biasScale = 1.f;
    la.out = act(0); la.numPreds = 1; la.K = 784;
    dag_layer_kernel<false, false, false><<<grid, blk, 0, stream>>>(la);
  }

  // DAG layers 1..5: accumulate all predecessor GEMMs in-register; K = 2048 aligned.
  int idx = 0;
  for (int i = 1; i < 6; ++i) {
    LayerArgs la{};
    for (int j = 0; j < i; ++j) {
      la.A[j] = act(j);
      la.W[j] = W + (size_t)(idx + j) * N_DIM * N_DIM;
      la.M[j] = M + (size_t)(idx + j) * N_DIM * N_DIM;
    }
    la.bias = bvec + (size_t)(i - 1) * N_DIM;
    la.biasScale = (float)i;
    la.out = act(i); la.numPreds = i; la.K = N_DIM;
    dag_layer_kernel<true, true, true><<<grid, blk, 0, stream>>>(la);
    idx += i;
  }

  classifier_kernel<<<B_DIM, 256, 0, stream>>>(act(5), act(4), w1, b1, w2, b2, out);
}
